// MultiHeadAttention_43722767073406
// MI455X (gfx1250) — matrix operand order repaired and checked by reading, not compiled, not run on hardware
//
#include <hip/hip_runtime.h>
#include <hip/hip_bf16.h>
#include <float.h>

typedef __attribute__((ext_vector_type(16))) __bf16 v16bf;
typedef __attribute__((ext_vector_type(8)))  __bf16 v8bf;
typedef __attribute__((ext_vector_type(4)))  __bf16 v4bf;
typedef __attribute__((ext_vector_type(8)))  float  v8f;
typedef __attribute__((ext_vector_type(4)))  float  v4f;

#define BATCH  2
#define SEQ    2048
#define DIM    1024
#define HEADS  16
#define DHEAD  64
#define INNER  1024
#define ROWS   (BATCH * SEQ)     /* 4096 */
#define ATT_SCALE 0.125f         /* 64^-0.5 */
#define NEGINF (-3.0e38f)

__device__ __forceinline__ v16bf combine16(v8bf lo, v8bf hi) {
  return __builtin_shufflevector(lo, hi, 0, 1, 2, 3, 4, 5, 6, 7,
                                         8, 9, 10, 11, 12, 13, 14, 15);
}

// Async DMA: copy 16 bytes per lane from global to LDS (ASYNCcnt-tracked).
// GV mode: LDS byte offset in one VGPR, 64-bit global address in a VGPR pair.
__device__ __forceinline__ void async_copy_b128(void* lds_dst, const void* gsrc) {
  unsigned int l = (unsigned int)(unsigned long long)lds_dst; // low 32b = LDS offset
  asm volatile("global_load_async_to_lds_b128 %0, %1, off"
               :: "v"(l), "v"((unsigned long long)gsrc)
               : "memory");
}
__device__ __forceinline__ void wait_async0() {
  asm volatile("s_wait_asynccnt 0" ::: "memory");
}

// A-fragment (16-bit 16x32): lane (row=lane%16, half hh) needs two contiguous
// 8-element runs at K = hh*8 and K = 16 + hh*8.
// B-fragment (16-bit 32x16): lane (col=lane%16, half hh) needs two contiguous
// 8-element runs at K = hh*8 and K = 16 + hh*8  -> stage B tiles [N][K] in LDS.

// ---------------------------------------------------------------------------
// Kernel 1: QKV projection.  Y = X[4096x1024] @ W[1024x1024] for W in
// {Wq,Wk,Wv} (blockIdx.z), output bf16 in [B,H,N,DHEAD] layout.
// Q output is pre-scaled by DHEAD^-0.5.
// ---------------------------------------------------------------------------
__global__ __launch_bounds__(128)
void qkv_proj_kernel(const float* __restrict__ x,
                     const float* __restrict__ Wq,
                     const float* __restrict__ Wk,
                     const float* __restrict__ Wv,
                     __bf16* __restrict__ Qb,
                     __bf16* __restrict__ Kb,
                     __bf16* __restrict__ Vb)
{
  const float* W  = (blockIdx.z == 0) ? Wq : (blockIdx.z == 1) ? Wk : Wv;
  __bf16*     Out = (blockIdx.z == 0) ? Qb : (blockIdx.z == 1) ? Kb : Vb;
  const float scl = (blockIdx.z == 0) ? ATT_SCALE : 1.0f;

  const int n0 = blockIdx.x * 64;
  const int m0 = blockIdx.y * 64;
  const int tid = threadIdx.x;
  const int wave = tid >> 5;
  const int lane = tid & 31;
  const int lane16 = lane & 15;
  const int hh = lane >> 4;

  __shared__ __bf16 Xs[64][40];   // A tile, row-major: [row][k]   (stride 80B)
  __shared__ __bf16 Wt[64][40];   // B tile, transposed: [col][k]  (stride 80B)

  v8f acc[4];
  #pragma unroll
  for (int t = 0; t < 4; ++t)
    #pragma unroll
    for (int r = 0; r < 8; ++r) acc[t][r] = 0.0f;

  for (int k0 = 0; k0 < DIM; k0 += 32) {
    #pragma unroll
    for (int e = 0; e < 4; ++e) {
      int flat = (e * 128 + tid) * 4;            // 0..2047, step 4
      int r = flat >> 5, c = flat & 31;
      v4f xv = *(const v4f*)&x[(size_t)(m0 + r) * DIM + (k0 + c)];
      v4bf xb;
      #pragma unroll
      for (int i = 0; i < 4; ++i) xb[i] = (__bf16)xv[i];
      *(v4bf*)&Xs[r][c] = xb;                    // ds_store_b64

      int r2 = flat >> 6, c2 = flat & 63;
      v4f wv = *(const v4f*)&W[(size_t)(k0 + r2) * INNER + (n0 + c2)];
      #pragma unroll
      for (int i = 0; i < 4; ++i) Wt[c2 + i][r2] = (__bf16)wv[i];
    }
    __syncthreads();

    v16bf a = combine16(*(const v8bf*)&Xs[wave * 16 + lane16][hh * 8],
                        *(const v8bf*)&Xs[wave * 16 + lane16][16 + hh * 8]);
    #pragma unroll
    for (int t = 0; t < 4; ++t) {
      v16bf bfr = combine16(*(const v8bf*)&Wt[t * 16 + lane16][hh * 8],
                            *(const v8bf*)&Wt[t * 16 + lane16][16 + hh * 8]);
      acc[t] = __builtin_amdgcn_wmma_f32_16x16x32_bf16(
          false, a, false, bfr, (short)0, acc[t], false, false);
    }
    __syncthreads();
  }

  // C layout: lane holds (M = r + 8*hh, N = lane16) for VGPR r.
  #pragma unroll
  for (int t = 0; t < 4; ++t) {
    int c = n0 + t * 16 + lane16;
    int h = c >> 6, d = c & 63;
    #pragma unroll
    for (int r = 0; r < 8; ++r) {
      int row = m0 + wave * 16 + r + 8 * hh;
      int b = row >> 11, n = row & (SEQ - 1);
      Out[((size_t)(b * HEADS + h) * SEQ + n) * DHEAD + d] =
          (__bf16)(acc[t][r] * scl);
    }
  }
}

// ---------------------------------------------------------------------------
// Kernel 2: causal flash attention.  One workgroup = (b*H+h, 64-query tile).
// Q/K tiles staged with async global->LDS DMA; V staged transposed manually.
// ---------------------------------------------------------------------------
__global__ __launch_bounds__(128)
void flash_attn_kernel(const __bf16* __restrict__ Qb,
                       const __bf16* __restrict__ Kb,
                       const __bf16* __restrict__ Vb,
                       const unsigned char* __restrict__ mask,
                       __bf16* __restrict__ Ob)
{
  const int bh = blockIdx.y;
  const int b = bh / HEADS, h = bh % HEADS;
  const int it = blockIdx.x;
  const int m0 = it * 64;
  const int tid = threadIdx.x;
  const int wave = tid >> 5;
  const int lane = tid & 31;
  const int lane16 = lane & 15;
  const int hh = lane >> 4;

  const __bf16* Qp = Qb + (size_t)bh * SEQ * DHEAD;
  const __bf16* Kp = Kb + (size_t)bh * SEQ * DHEAD;
  const __bf16* Vp = Vb + (size_t)bh * SEQ * DHEAD;

  __shared__ __bf16 Qs[64][72];        // [query][d]   row-major  (stride 144B)
  __shared__ __bf16 Ks[64][72];        // [key][d]     row-major (B of Q*K^T)
  __shared__ __bf16 Vt[64][72];        // [d][key]     transposed (B of P*V)
  __shared__ __bf16 Ps[4][16][72];     // per-wave P staging, row-major [q][key]

  #pragma unroll
  for (int e = 0; e < 4; ++e) {
    int flat = (e * 128 + tid) * 8;              // 0..4095, step 8
    int r = flat >> 6, d = flat & 63;
    async_copy_b128(&Qs[r][d], &Qp[(size_t)(m0 + r) * DHEAD + d]);
  }

  float mrow[8], lrow[8];
  v8f oacc[4];
  #pragma unroll
  for (int r = 0; r < 8; ++r) { mrow[r] = -FLT_MAX; lrow[r] = 0.0f; }
  #pragma unroll
  for (int t = 0; t < 4; ++t)
    #pragma unroll
    for (int r = 0; r < 8; ++r) oacc[t][r] = 0.0f;

  for (int jb = 0; jb <= it; ++jb) {
    const int j0 = jb * 64;
    __syncthreads();                             // Ks/Vt safe to overwrite
    #pragma unroll
    for (int e = 0; e < 4; ++e) {
      int flat = (e * 128 + tid) * 8;
      int r = flat >> 6, d = flat & 63;
      async_copy_b128(&Ks[r][d], &Kp[(size_t)(j0 + r) * DHEAD + d]);
      v8bf vv = *(const v8bf*)&Vp[(size_t)(j0 + r) * DHEAD + d];
      #pragma unroll
      for (int i = 0; i < 8; ++i) Vt[d + i][r] = vv[i];
    }
    wait_async0();                               // Q (iter 0) + K DMA done
    __syncthreads();

    if (jb < it) {                               // prefetch next K/V block
      __builtin_prefetch(&Kp[(size_t)(j0 + 64) * DHEAD + lane * 16], 0, 1);
      __builtin_prefetch(&Vp[(size_t)(j0 + 64) * DHEAD + lane * 16], 0, 1);
    }

    // ---- S = Q * K^T  (K-dim = DHEAD = 64 -> 2 WMMA k-steps) ----
    v8f s[4];
    #pragma unroll
    for (int t = 0; t < 4; ++t)
      #pragma unroll
      for (int r = 0; r < 8; ++r) s[t][r] = 0.0f;

    #pragma unroll
    for (int ks = 0; ks < 2; ++ks) {
      v16bf a = combine16(
          *(const v8bf*)&Qs[wave * 16 + lane16][ks * 32 + hh * 8],
          *(const v8bf*)&Qs[wave * 16 + lane16][ks * 32 + 16 + hh * 8]);
      #pragma unroll
      for (int t = 0; t < 4; ++t) {
        // B(k=d, n=key) = K[key][d]: contiguous along d in Ks rows.
        v16bf bfr = combine16(
            *(const v8bf*)&Ks[t * 16 + lane16][ks * 32 + hh * 8],
            *(const v8bf*)&Ks[t * 16 + lane16][ks * 32 + 16 + hh * 8]);
        s[t] = __builtin_amdgcn_wmma_f32_16x16x32_bf16(
            false, a, false, bfr, (short)0, s[t], false, false);
      }
    }

    // ---- causal / padding mask (scale pre-folded into Q) ----
    #pragma unroll
    for (int t = 0; t < 4; ++t) {
      int jg = j0 + t * 16 + lane16;
      bool mok = mask[(size_t)b * SEQ + jg] != 0;
      #pragma unroll
      for (int r = 0; r < 8; ++r) {
        int ig = m0 + wave * 16 + r + 8 * hh;
        if ((jg > ig) || !mok) s[t][r] = NEGINF;
      }
    }

    // ---- online softmax (row reductions across one 16-lane half-group) ----
    #pragma unroll
    for (int r = 0; r < 8; ++r) {
      float tmax = s[0][r];
      #pragma unroll
      for (int t = 1; t < 4; ++t) tmax = fmaxf(tmax, s[t][r]);
      #pragma unroll
      for (int off = 1; off < 16; off <<= 1)
        tmax = fmaxf(tmax, __shfl_xor(tmax, off, 32));
      float mnew = fmaxf(mrow[r], tmax);
      float fac  = __expf(mrow[r] - mnew);
      float sum = 0.0f;
      #pragma unroll
      for (int t = 0; t < 4; ++t) {
        float p = __expf(s[t][r] - mnew);
        s[t][r] = p;
        sum += p;
      }
      #pragma unroll
      for (int off = 1; off < 16; off <<= 1)
        sum += __shfl_xor(sum, off, 32);
      lrow[r] = lrow[r] * fac + sum;
      mrow[r] = mnew;
      #pragma unroll
      for (int t = 0; t < 4; ++t) oacc[t][r] *= fac;
    }

    // ---- re-stripe P: C layout -> A-fragment layout via per-wave LDS ----
    #pragma unroll
    for (int t = 0; t < 4; ++t)
      #pragma unroll
      for (int r = 0; r < 8; ++r)
        Ps[wave][r + 8 * hh][t * 16 + lane16] = (__bf16)s[t][r];

    // ---- O += P @ V  (K-dim = 64 keys -> 2 WMMA k-steps) ----
    #pragma unroll
    for (int ks = 0; ks < 2; ++ks) {
      v16bf a = combine16(
          *(const v8bf*)&Ps[wave][lane16][ks * 32 + hh * 8],
          *(const v8bf*)&Ps[wave][lane16][ks * 32 + 16 + hh * 8]);
      #pragma unroll
      for (int t = 0; t < 4; ++t) {
        // B(k=key, n=d) = V[key][d] = Vt[d][key]: contiguous along key.
        v16bf bfr = combine16(
            *(const v8bf*)&Vt[t * 16 + lane16][ks * 32 + hh * 8],
            *(const v8bf*)&Vt[t * 16 + lane16][ks * 32 + 16 + hh * 8]);
        oacc[t] = __builtin_amdgcn_wmma_f32_16x16x32_bf16(
            false, a, false, bfr, (short)0, oacc[t], false, false);
      }
    }
  }

  // ---- normalize and store O as [B, N, H*DHEAD] bf16 ----
  #pragma unroll
  for (int r = 0; r < 8; ++r) {
    int i = m0 + wave * 16 + r + 8 * hh;
    float inv = 1.0f / fmaxf(lrow[r], 1e-30f);
    #pragma unroll
    for (int t = 0; t < 4; ++t)
      Ob[((size_t)b * SEQ + i) * INNER + h * DHEAD + t * 16 + lane16] =
          (__bf16)(oacc[t][r] * inv);
  }
}

// ---------------------------------------------------------------------------
// Kernel 3: output projection.  out = O[4096x1024]bf16 @ Wo[1024x1024] -> f32
// A tile staged via async DMA (already bf16); W converted f32->bf16 manually.
// ---------------------------------------------------------------------------
__global__ __launch_bounds__(128)
void out_proj_kernel(const __bf16* __restrict__ A,
                     const float* __restrict__ Wo,
                     float* __restrict__ Out)
{
  const int n0 = blockIdx.x * 64;
  const int m0 = blockIdx.y * 64;
  const int tid = threadIdx.x;
  const int wave = tid >> 5;
  const int lane = tid & 31;
  const int lane16 = lane & 15;
  const int hh = lane >> 4;

  __shared__ __bf16 Xs[64][40];   // A tile row-major
  __shared__ __bf16 Wt[64][40];   // B tile transposed [col][k]

  v8f acc[4];
  #pragma unroll
  for (int t = 0; t < 4; ++t)
    #pragma unroll
    for (int r = 0; r < 8; ++r) acc[t][r] = 0.0f;

  for (int k0 = 0; k0 < INNER; k0 += 32) {
    #pragma unroll
    for (int e = 0; e < 2; ++e) {
      int flat = (e * 128 + tid) * 8;            // A: 0..2047, step 8 (bf16)
      int r = flat >> 5, c = flat & 31;
      async_copy_b128(&Xs[r][c], &A[(size_t)(m0 + r) * INNER + (k0 + c)]);
    }
    #pragma unroll
    for (int e = 0; e < 4; ++e) {
      int flat = (e * 128 + tid) * 4;            // W: 0..2047, step 4 (f32)
      int r2 = flat >> 6, c2 = flat & 63;
      v4f wv = *(const v4f*)&Wo[(size_t)(k0 + r2) * DIM + (n0 + c2)];
      #pragma unroll
      for (int i = 0; i < 4; ++i) Wt[c2 + i][r2] = (__bf16)wv[i];
    }
    wait_async0();
    __syncthreads();

    v16bf a = combine16(*(const v8bf*)&Xs[wave * 16 + lane16][hh * 8],
                        *(const v8bf*)&Xs[wave * 16 + lane16][16 + hh * 8]);
    #pragma unroll
    for (int t = 0; t < 4; ++t) {
      v16bf bfr = combine16(*(const v8bf*)&Wt[t * 16 + lane16][hh * 8],
                            *(const v8bf*)&Wt[t * 16 + lane16][16 + hh * 8]);
      acc[t] = __builtin_amdgcn_wmma_f32_16x16x32_bf16(
          false, a, false, bfr, (short)0, acc[t], false, false);
    }
    __syncthreads();
  }

  #pragma unroll
  for (int t = 0; t < 4; ++t)
    #pragma unroll
    for (int r = 0; r < 8; ++r) {
      int row = m0 + wave * 16 + r + 8 * hh;
      Out[(size_t)row * DIM + n0 + t * 16 + lane16] = acc[t][r];
    }
}

// ---------------------------------------------------------------------------
extern "C" void kernel_launch(void* const* d_in, const int* in_sizes, int n_in,
                              void* d_out, int out_size, void* d_ws, size_t ws_size,
                              hipStream_t stream) {
  (void)in_sizes; (void)n_in; (void)out_size; (void)ws_size;
  const float* x          = (const float*)d_in[0];
  const unsigned char* mk = (const unsigned char*)d_in[1];  // bool mask
  const float* Wq         = (const float*)d_in[2];
  const float* Wk         = (const float*)d_in[3];
  const float* Wv         = (const float*)d_in[4];
  const float* Wo         = (const float*)d_in[5];

  const size_t elems = (size_t)BATCH * HEADS * SEQ * DHEAD;  // 4M bf16 each
  char* ws = (char*)d_ws;
  __bf16* Qb = (__bf16*)(ws);
  __bf16* Kb = (__bf16*)(ws + 2 * elems);
  __bf16* Vb = (__bf16*)(ws + 4 * elems);
  __bf16* Ob = (__bf16*)(ws + 6 * elems);

  dim3 blk(128);
  dim3 g1(INNER / 64, ROWS / 64, 3);
  qkv_proj_kernel<<<g1, blk, 0, stream>>>(x, Wq, Wk, Wv, Qb, Kb, Vb);

  dim3 g2(SEQ / 64, BATCH * HEADS);
  flash_attn_kernel<<<g2, blk, 0, stream>>>(Qb, Kb, Vb, mk, Ob);

  dim3 g3(DIM / 64, ROWS / 64);
  out_proj_kernel<<<g3, blk, 0, stream>>>(Ob, Wo, (float*)d_out);
}
